// Attention_9715216023711
// MI455X (gfx1250) — matrix operand order repaired and checked by reading, not compiled, not run on hardware
//
#include <hip/hip_runtime.h>
#include <hip/hip_bf16.h>

// ---------------------------------------------------------------------------
// CDNA5 (gfx1250) wave32 WMMA attention:  out = proj(softmax(QK^T/sqrt(D)) V)
// All matmuls use v_wmma_f32_16x16x32_bf16 (fp32 accumulate).
// ---------------------------------------------------------------------------

typedef __attribute__((ext_vector_type(16))) __bf16 bfrag;   // A/B fragment: 8 VGPRs
typedef __attribute__((ext_vector_type(8)))  float  ffrag;   // C/D fragment: 8 VGPRs

// WMMA wrapper: D = A(16x32 bf16) * B(32x16 bf16) + C(16x16 f32)
__device__ __forceinline__ ffrag wmma_bf16(bfrag a, bfrag b, ffrag c) {
  return __builtin_amdgcn_wmma_f32_16x16x32_bf16(
      /*neg_a=*/false, a, /*neg_b=*/false, b,
      /*c_mod=*/(short)0, c, /*reuse_a=*/false, /*reuse_b=*/false);
}

// A fragment: 16(M) x 32(K) bf16 tile from row-major src (ld elems/row).
// ISA layout (CDNA5 ISA): lane&15 = M; elements 0..7 -> K = 8*hi+e,
// elements 8..15 -> K = 16 + 8*hi + e  (hi = lane>>4). Two 16B loads per lane.
__device__ __forceinline__ bfrag load_a(const __bf16* src, int ld) {
  const int lane = threadIdx.x & 31;
  const int m = lane & 15, hi = lane >> 4;
  const __bf16* p = src + (long)m * ld + hi * 8;
  union { bfrag v; uint4 q[2]; } u;
  u.q[0] = *(const uint4*)(p);
  u.q[1] = *(const uint4*)(p + 16);
  return u.v;
}

// B fragment: 32(K) x 16(N) bf16 where B = T^T, T row-major with leading dim ld
// (row n of T is column n of B).  Lane layout: n = lane&15, elements 0..7 ->
// K = 8*hi+e, elements 8..15 -> K = 16 + 8*hi + e. Two 16B loads per lane.
__device__ __forceinline__ bfrag load_b(const __bf16* t, int ld) {
  const int lane = threadIdx.x & 31;
  const int n = lane & 15, hi = lane >> 4;
  const __bf16* p = t + (long)n * ld + hi * 8;
  union { bfrag v; uint4 q[2]; } u;
  u.q[0] = *(const uint4*)(p);
  u.q[1] = *(const uint4*)(p + 16);
  return u.v;
}

// ---------------------------------------------------------------------------
// Prep kernels: fp32 -> bf16 convert, and transposed bf16 weight copies
// ---------------------------------------------------------------------------
__global__ void cvt_bf16_kernel(const float* __restrict__ in, __bf16* __restrict__ out, int n) {
  int i = blockIdx.x * blockDim.x + threadIdx.x;
  if (i < n) out[i] = (__bf16)in[i];
}

// in: [rows][cols] fp32 -> out: [cols][rows] bf16
__global__ void transpose_bf16_kernel(const float* __restrict__ in, __bf16* __restrict__ out,
                                      int rows, int cols) {
  int o = blockIdx.x * blockDim.x + threadIdx.x;
  if (o >= rows * cols) return;
  int on = o / rows;        // output row  = input col
  int ok = o - on * rows;   // output col  = input row
  out[o] = (__bf16)in[(long)ok * cols + on];
}

// ---------------------------------------------------------------------------
// QKV GEMM: [4096 x 1024] @ [1024 x 3072] -> split into Q,K,V [BH][2048][64]
// bf16, with softmax scale (1/8) folded into Q.
// Wave tile = 64x64 (16 WMMAs per K32 step, 16 b128 loads -> 1.0 load/WMMA).
// __launch_bounds__(128,1): ~212 VGPRs/wave, no spills (gfx1250 allows 1024).
// ---------------------------------------------------------------------------
__global__ void __launch_bounds__(128, 1)
qkv_gemm_kernel(const __bf16* __restrict__ xb, const __bf16* __restrict__ wT,
                __bf16* __restrict__ qo, __bf16* __restrict__ ko,
                __bf16* __restrict__ vo) {
  const int wave = threadIdx.x >> 5;
  const int lane = threadIdx.x & 31;
  const int n16 = lane & 15, hi = lane >> 4;
  const int row0 = blockIdx.x * 64;
  const int col0 = blockIdx.y * 256 + wave * 64;

  ffrag acc[4][4];
  #pragma unroll
  for (int i = 0; i < 4; ++i)
    #pragma unroll
    for (int j = 0; j < 4; ++j) acc[i][j] = ffrag{};

  for (int k0 = 0; k0 < 1024; k0 += 32) {
    bfrag af[4], bf[4];
    #pragma unroll
    for (int i = 0; i < 4; ++i) af[i] = load_a(xb + (long)(row0 + 16 * i) * 1024 + k0, 1024);
    #pragma unroll
    for (int j = 0; j < 4; ++j) bf[j] = load_b(wT + (long)(col0 + 16 * j) * 1024 + k0, 1024);
    if (k0 + 32 < 1024) {
      __builtin_prefetch(xb + (long)row0 * 1024 + k0 + 32, 0, 3);  // WGP-scope
      __builtin_prefetch(wT + (long)col0 * 1024 + k0 + 32, 0, 3);
    }
    #pragma unroll
    for (int i = 0; i < 4; ++i)
      #pragma unroll
      for (int j = 0; j < 4; ++j) acc[i][j] = wmma_bf16(af[i], bf[j], acc[i][j]);
  }

  // Epilogue: scatter into per-head [BH][2048][64] Q/K/V (scale folded into Q).
  #pragma unroll
  for (int i = 0; i < 4; ++i) {
    #pragma unroll
    for (int j = 0; j < 4; ++j) {
      #pragma unroll
      for (int r = 0; r < 8; ++r) {
        int m = row0 + 16 * i + r + 8 * hi;   // row in [0,4096)
        int c = col0 + 16 * j + n16;          // col in [0,3072)
        float val = acc[i][j][r];
        int sect = c >> 10;                   // 0=Q 1=K 2=V
        int cc = c & 1023;
        int h = cc >> 6, d = cc & 63;
        int bb = m >> 11, s = m & 2047;
        long idx = (((long)(bb * 16 + h)) * 2048 + s) * 64 + d;
        if (sect == 0)      qo[idx] = (__bf16)(val * 0.125f);
        else if (sect == 1) ko[idx] = (__bf16)val;
        else                vo[idx] = (__bf16)val;
      }
    }
  }
}

// ---------------------------------------------------------------------------
// Flash attention: grid (BH=32, N/64=32), 4 waves x 16 query rows each.
// Key tiles of 32: K staged in LDS as-is, V staged transposed; per-wave LDS
// round-trip converts P from the C/D layout to the A layout.
// ---------------------------------------------------------------------------
__global__ void flash_attn_kernel(const __bf16* __restrict__ qg, const __bf16* __restrict__ kg,
                                  const __bf16* __restrict__ vg, __bf16* __restrict__ out) {
  __shared__ __align__(16) __bf16 Ks[32][64];      // key tile  [key][d]
  __shared__ __align__(16) __bf16 Vt[64][32];      // val tile  [d][key] (transposed)
  __shared__ __align__(16) __bf16 Pb[4][16][32];   // per-wave P staging

  const int wave = threadIdx.x >> 5;
  const int lane = threadIdx.x & 31;
  const int n16 = lane & 15, hi = lane >> 4;
  const int bh = blockIdx.x;                       // 0..31
  const int b = bh >> 4, h = bh & 15;
  const int q0 = blockIdx.y * 64 + wave * 16;
  const long base = (long)bh * 2048 * 64;

  // Q fragments (scale already folded in), resident for the whole kernel.
  const bfrag qa0 = load_a(qg + base + (long)q0 * 64 + 0, 64);
  const bfrag qa1 = load_a(qg + base + (long)q0 * 64 + 32, 64);

  ffrag o0 = {}, o1 = {}, o2 = {}, o3 = {};        // O accum, d-chunks of 16
  float m_[8], l_[8];
  #pragma unroll
  for (int r = 0; r < 8; ++r) { m_[r] = -1e30f; l_[r] = 0.f; }

  const __bf16* Kb = kg + base;
  const __bf16* Vb = vg + base;

  for (int kk = 0; kk < 2048; kk += 32) {
    // Cooperative staging: 32x64 bf16 of K and V (V stored transposed).
    #pragma unroll
    for (int it = 0; it < 2; ++it) {
      int id = threadIdx.x + it * 128;             // 0..255 16B-chunks
      int row = id >> 3;
      int c8 = (id & 7) * 8;
      *(uint4*)(&Ks[row][c8]) = *(const uint4*)(Kb + (long)(kk + row) * 64 + c8);
      union { uint4 u; __bf16 e[8]; } vv;
      vv.u = *(const uint4*)(Vb + (long)(kk + row) * 64 + c8);
      #pragma unroll
      for (int j = 0; j < 8; ++j) Vt[c8 + j][row] = vv.e[j];
    }
    __syncthreads();

    // S = Q * K^T : two 16x16 key tiles, K-dim 64 = 2 WMMAs each.
    ffrag s0 = {}, s1 = {};
    s0 = wmma_bf16(qa0, load_b(&Ks[0][0], 64), s0);
    s0 = wmma_bf16(qa1, load_b(&Ks[0][32], 64), s0);
    s1 = wmma_bf16(qa0, load_b(&Ks[16][0], 64), s1);
    s1 = wmma_bf16(qa1, load_b(&Ks[16][32], 64), s1);

    // Online softmax. Row r+8*hi stats live replicated in each 16-lane half.
    float alpha[8];
    #pragma unroll
    for (int r = 0; r < 8; ++r) {
      float cmax = fmaxf(s0[r], s1[r]);
      #pragma unroll
      for (int off = 1; off < 16; off <<= 1) cmax = fmaxf(cmax, __shfl_xor(cmax, off, 32));
      float mn = fmaxf(m_[r], cmax);
      alpha[r] = __expf(m_[r] - mn);
      float p0 = __expf(s0[r] - mn);
      float p1 = __expf(s1[r] - mn);
      float rs = p0 + p1;
      #pragma unroll
      for (int off = 1; off < 16; off <<= 1) rs += __shfl_xor(rs, off, 32);
      l_[r] = alpha[r] * l_[r] + rs;
      m_[r] = mn;
      s0[r] = p0;
      s1[r] = p1;
    }
    #pragma unroll
    for (int r = 0; r < 8; ++r) {
      o0[r] *= alpha[r]; o1[r] *= alpha[r]; o2[r] *= alpha[r]; o3[r] *= alpha[r];
    }

    // P: C/D layout -> A layout via per-wave LDS round-trip (LDS is in-order
    // per wave; explicit dscnt drain before the cross-lane read-back).
    __bf16* pp = &Pb[wave][0][0];
    #pragma unroll
    for (int r = 0; r < 8; ++r) {
      pp[(r + 8 * hi) * 32 + n16]      = (__bf16)s0[r];
      pp[(r + 8 * hi) * 32 + 16 + n16] = (__bf16)s1[r];
    }
    asm volatile("s_wait_dscnt 0" ::: "memory");
    bfrag pa = load_a(pp, 32);

    // O += P(16x32) * V(32x64): 4 WMMAs (one per 16-wide d chunk).
    o0 = wmma_bf16(pa, load_b(&Vt[0][0], 32), o0);
    o1 = wmma_bf16(pa, load_b(&Vt[16][0], 32), o1);
    o2 = wmma_bf16(pa, load_b(&Vt[32][0], 32), o2);
    o3 = wmma_bf16(pa, load_b(&Vt[48][0], 32), o3);

    if (kk + 32 < 2048) {
      __builtin_prefetch(Kb + (long)(kk + 32) * 64, 0, 3);
      __builtin_prefetch(Vb + (long)(kk + 32) * 64, 0, 3);
    }
    __syncthreads();
  }

  // Normalize and write bf16 [B][N][C] (head-concat layout for the proj GEMM).
  #pragma unroll
  for (int r = 0; r < 8; ++r) {
    float inv = 1.0f / l_[r];
    int nrow = q0 + r + 8 * hi;
    long o = ((long)(b * 2048 + nrow)) * 1024 + h * 64;
    out[o + 0  + n16] = (__bf16)(o0[r] * inv);
    out[o + 16 + n16] = (__bf16)(o1[r] * inv);
    out[o + 32 + n16] = (__bf16)(o2[r] * inv);
    out[o + 48 + n16] = (__bf16)(o3[r] * inv);
  }
}

// ---------------------------------------------------------------------------
// Projection GEMM: [4096 x 1024] @ [1024 x 1024] + bias -> fp32 d_out
// Wave tile 64x64, block covers 64 x 256.
// ---------------------------------------------------------------------------
__global__ void __launch_bounds__(128, 1)
proj_gemm_kernel(const __bf16* __restrict__ ab, const __bf16* __restrict__ wT,
                 const float* __restrict__ bias, float* __restrict__ out) {
  const int wave = threadIdx.x >> 5;
  const int lane = threadIdx.x & 31;
  const int n16 = lane & 15, hi = lane >> 4;
  const int row0 = blockIdx.x * 64;
  const int col0 = blockIdx.y * 256 + wave * 64;

  ffrag acc[4][4];
  #pragma unroll
  for (int i = 0; i < 4; ++i)
    #pragma unroll
    for (int j = 0; j < 4; ++j) acc[i][j] = ffrag{};

  for (int k0 = 0; k0 < 1024; k0 += 32) {
    bfrag af[4], bf[4];
    #pragma unroll
    for (int i = 0; i < 4; ++i) af[i] = load_a(ab + (long)(row0 + 16 * i) * 1024 + k0, 1024);
    #pragma unroll
    for (int j = 0; j < 4; ++j) bf[j] = load_b(wT + (long)(col0 + 16 * j) * 1024 + k0, 1024);
    if (k0 + 32 < 1024) {
      __builtin_prefetch(ab + (long)row0 * 1024 + k0 + 32, 0, 3);
      __builtin_prefetch(wT + (long)col0 * 1024 + k0 + 32, 0, 3);
    }
    #pragma unroll
    for (int i = 0; i < 4; ++i)
      #pragma unroll
      for (int j = 0; j < 4; ++j) acc[i][j] = wmma_bf16(af[i], bf[j], acc[i][j]);
  }

  #pragma unroll
  for (int i = 0; i < 4; ++i) {
    #pragma unroll
    for (int j = 0; j < 4; ++j) {
      #pragma unroll
      for (int r = 0; r < 8; ++r) {
        int m = row0 + 16 * i + r + 8 * hi;
        int c = col0 + 16 * j + n16;
        out[(long)m * 1024 + c] = acc[i][j][r] + bias[c];
      }
    }
  }
}

// ---------------------------------------------------------------------------
// Host launcher
// ---------------------------------------------------------------------------
extern "C" void kernel_launch(void* const* d_in, const int* in_sizes, int n_in,
                              void* d_out, int out_size, void* d_ws, size_t ws_size,
                              hipStream_t stream) {
  (void)in_sizes; (void)n_in; (void)out_size; (void)ws_size;
  const float* x      = (const float*)d_in[0];   // [2,2048,1024]
  const float* w_qkv  = (const float*)d_in[1];   // [1024,3072]
  const float* w_proj = (const float*)d_in[2];   // [1024,1024]
  const float* b_proj = (const float*)d_in[3];   // [1024]
  float* out = (float*)d_out;

  char* ws = (char*)d_ws;
  __bf16* xb     = (__bf16*)(ws + 0);          //  8 MB  x bf16        [4096,1024]
  __bf16* wqkvT  = (__bf16*)(ws + 8388608);    //  6 MB  w_qkv^T bf16  [3072,1024]
  __bf16* wprojT = (__bf16*)(ws + 14680064);   //  2 MB  w_proj^T bf16 [1024,1024]
  __bf16* qb     = (__bf16*)(ws + 16777216);   //  8 MB  Q bf16 [32][2048][64] (scaled)
  __bf16* kb     = (__bf16*)(ws + 25165824);   //  8 MB  K bf16
  __bf16* vb     = (__bf16*)(ws + 33554432);   //  8 MB  V bf16
  __bf16* att    = (__bf16*)(ws + 41943040);   //  8 MB  attn out bf16 [4096,1024]

  cvt_bf16_kernel<<<4194304 / 256, 256, 0, stream>>>(x, xb, 4194304);
  transpose_bf16_kernel<<<3145728 / 256, 256, 0, stream>>>(w_qkv, wqkvT, 1024, 3072);
  transpose_bf16_kernel<<<1048576 / 256, 256, 0, stream>>>(w_proj, wprojT, 1024, 1024);

  qkv_gemm_kernel<<<dim3(64, 12), 128, 0, stream>>>(xb, wqkvT, qb, kb, vb);
  flash_attn_kernel<<<dim3(32, 32), 128, 0, stream>>>(qb, kb, vb, att);
  proj_gemm_kernel<<<dim3(64, 4), 128, 0, stream>>>(att, wprojT, b_proj, out);
}
